// ElmanRNN_24283745091918
// MI455X (gfx1250) — matrix operand order repaired and checked by reading, not compiled, not run on hardware
//
#include <hip/hip_runtime.h>
#include <hip/hip_bf16.h>
#include <math.h>

// ---------------------------------------------------------------------------
// Elman RNN on MI455X (gfx1250, wave32, WMMA).
//   h_t = tanh(x_t @ W + h_{t-1} @ U + b),  out = h_T @ Wf + bf
// B=128, T=512, F=256, H=512.
//
// The scan is latency-bound (67 MFLOP/step), so the input projection is
// folded into each step (K = 256 + 512) and steps run as 512 stream-ordered
// launches over ping-pong h buffers. All WMMA operands are bf16 in memory
// (weights pre-transposed, x pre-converted, h stored as bf16), so the hot
// loop is pure b128 loads + v_wmma_f32_16x16x32_bf16. Two accumulators break
// the serial D->C WMMA chain into two interleaved chains.
// ---------------------------------------------------------------------------

#define RNN_B 128
#define RNN_T 512
#define RNN_F 256
#define RNN_H 512

typedef __attribute__((ext_vector_type(16))) __bf16 v16bf;
typedef __attribute__((ext_vector_type(8)))  float  v8f;

__device__ __forceinline__ unsigned short f2bf_bits(float f) {
    unsigned u = __float_as_uint(f);
    u += 0x7FFFu + ((u >> 16) & 1u);        // round-to-nearest-even
    return (unsigned short)(u >> 16);
}
__device__ __forceinline__ float bf_bits2f(unsigned short s) {
    return __uint_as_float((unsigned)s << 16);
}

// 32-byte fragment viewed either as two uint4 (for b128 loads) or as the
// WMMA bf16 operand. Memory packing (2 bf16 per dword, little-endian) matches
// the v16bf element order exactly, so this is a pure reinterpret.
union Frag {
    uint4 q[2];
    v16bf v;
    unsigned short s[16];
};

// ---------------------------------------------------------------------------
// Kernel 0: weight prep (re-run every call for graph-replay determinism):
//   Wt[n*F + k] = bf16(W[k*H + n])   (W^T, bf16)   512*256
//   Ut[n*H + k] = bf16(U[k*H + n])   (U^T, bf16)   512*512
//   h0[:] = 0 (bf16)                                128*512
// ---------------------------------------------------------------------------
__global__ __launch_bounds__(256) void rnn_prep_kernel(
    const float* __restrict__ W, const float* __restrict__ U,
    unsigned short* __restrict__ Wt, unsigned short* __restrict__ Ut,
    unsigned short* __restrict__ h0)
{
    int idx = blockIdx.x * 256 + threadIdx.x;
    if (idx < RNN_H * RNN_F) {                      // W^T
        int n = idx / RNN_F, k = idx % RNN_F;
        Wt[idx] = f2bf_bits(W[(size_t)k * RNN_H + n]);
    }
    int j = idx - RNN_H * RNN_F;
    if (j >= 0 && j < RNN_H * RNN_H) {              // U^T
        int n = j / RNN_H, k = j % RNN_H;
        Ut[j] = f2bf_bits(U[(size_t)k * RNN_H + n]);
    }
    if (idx < RNN_B * RNN_H) h0[idx] = 0;           // h_0 = 0
}

// Kernel 0b: bulk x fp32 -> bf16 (coalesced, one-time, outside the scan).
__global__ __launch_bounds__(256) void rnn_xcvt_kernel(
    const float* __restrict__ x, unsigned short* __restrict__ xb, int n)
{
    int idx = blockIdx.x * 256 + threadIdx.x;
    if (idx < n) xb[idx] = f2bf_bits(x[idx]);
}

__device__ __forceinline__ v16bf load_afrag(const unsigned short* rowk, int lh) {
    Frag f;
    const unsigned short* p = rowk + lh * 8;
    f.q[0] = *(const uint4*)(p);        // K klo .. klo+7
    f.q[1] = *(const uint4*)(p + 16);   // K klo+16 .. klo+23
    return f.v;
}
__device__ __forceinline__ v16bf load_bfrag(const unsigned short* rowk, int lh) {
    Frag f;
    const unsigned short* p = rowk + lh * 8;
    f.q[0] = *(const uint4*)(p);        // K klo .. klo+7
    f.q[1] = *(const uint4*)(p + 16);   // K klo+16 .. klo+23
    return f.v;
}

// ---------------------------------------------------------------------------
// Per-timestep kernel: hn = tanh(x_t @ W + hp @ U + bias)   (hn, hp in bf16)
// Grid: 32 blocks (N tiles of 16 cols) x 256 threads (8 waves = 8 M tiles).
//
// WMMA wave32 layouts (CDNA5 ISA 7.12.2):
//   A 16x32 bf16 : row = lane%16; lanes<16 hold K {0..7,16..23},
//                  lanes>=16 hold K {8..15,24..31}  -> loads at +lh*8, +16.
//   B 32x16 bf16 : col = lane%16; K split same as A ({0..7,16..23} / {8..15,24..31})
//                  -> loads at +lh*8, +16.
//   C/D 16x16 f32: VGPR r -> M = r + 8*(lane>=16), N = lane%16.
// ---------------------------------------------------------------------------
template<bool PREX>
__global__ __launch_bounds__(256) void rnn_step_kernel(
    const float*          __restrict__ x,     // [B,T,F] fp32 (used if !PREX)
    const unsigned short* __restrict__ xb,    // [B,T,F] bf16 (used if PREX)
    const unsigned short* __restrict__ Wt,    // [H][F]  bf16 (W^T)
    const unsigned short* __restrict__ Ut,    // [H][H]  bf16 (U^T)
    const float*          __restrict__ bias,  // [H]
    const unsigned short* __restrict__ hp,    // [B][H] bf16 (h_{t-1})
    unsigned short*       __restrict__ hn,    // [B][H] bf16 (h_t)
    int t)
{
    const int lane = threadIdx.x & 31;
    const int wave = threadIdx.x >> 5;        // 0..7  -> M tile
    const int n0   = blockIdx.x * 16;         // N tile base
    const int m0   = wave * 16;               // M tile base (batch rows)
    const int rlo  = lane & 15;
    const int lh   = lane >> 4;

    // Two accumulators -> two independent WMMA dependency chains.
    v8f acc0 = {};
    v8f acc1 = {};

    // ---- Phase 1: x_t @ W   (K = F = 256) ----
    {
        const unsigned short* brow = Wt + (size_t)(n0 + rlo) * RNN_F;
        if (PREX) {
            const unsigned short* arow =
                xb + ((size_t)(m0 + rlo) * RNN_T + t) * RNN_F;
            #pragma unroll 2
            for (int k = 0; k < RNN_F; k += 64) {
                v16bf a0 = load_afrag(arow + k,      lh);
                v16bf b0 = load_bfrag(brow + k,      lh);
                v16bf a1 = load_afrag(arow + k + 32, lh);
                v16bf b1 = load_bfrag(brow + k + 32, lh);
                acc0 = __builtin_amdgcn_wmma_f32_16x16x32_bf16(
                    false, a0, false, b0, (short)0, acc0, false, false);
                acc1 = __builtin_amdgcn_wmma_f32_16x16x32_bf16(
                    false, a1, false, b1, (short)0, acc1, false, false);
            }
        } else {
            const float* arow = x + ((size_t)(m0 + rlo) * RNN_T + t) * RNN_F;
            #pragma unroll 2
            for (int k = 0; k < RNN_F; k += 64) {
                Frag a0, a1;
                const float* ap0 = arow + k + lh * 8;
                const float* ap1 = ap0 + 32;
                #pragma unroll
                for (int e = 0; e < 8; ++e) {
                    a0.s[e]     = f2bf_bits(ap0[e]);
                    a0.s[8 + e] = f2bf_bits(ap0[16 + e]);
                    a1.s[e]     = f2bf_bits(ap1[e]);
                    a1.s[8 + e] = f2bf_bits(ap1[16 + e]);
                }
                v16bf b0 = load_bfrag(brow + k,      lh);
                v16bf b1 = load_bfrag(brow + k + 32, lh);
                acc0 = __builtin_amdgcn_wmma_f32_16x16x32_bf16(
                    false, a0.v, false, b0, (short)0, acc0, false, false);
                acc1 = __builtin_amdgcn_wmma_f32_16x16x32_bf16(
                    false, a1.v, false, b1, (short)0, acc1, false, false);
            }
        }
    }

    // ---- Phase 2: h_{t-1} @ U   (K = H = 512), all-bf16 loads ----
    {
        const unsigned short* arow = hp + (size_t)(m0 + rlo) * RNN_H;
        const unsigned short* brow = Ut + (size_t)(n0 + rlo) * RNN_H;
        #pragma unroll 4
        for (int k = 0; k < RNN_H; k += 64) {
            v16bf a0 = load_afrag(arow + k,      lh);
            v16bf b0 = load_bfrag(brow + k,      lh);
            v16bf a1 = load_afrag(arow + k + 32, lh);
            v16bf b1 = load_bfrag(brow + k + 32, lh);
            acc0 = __builtin_amdgcn_wmma_f32_16x16x32_bf16(
                false, a0, false, b0, (short)0, acc0, false, false);
            acc1 = __builtin_amdgcn_wmma_f32_16x16x32_bf16(
                false, a1, false, b1, (short)0, acc1, false, false);
        }
    }

    // ---- merge chains, bias + tanh -> bf16 store (C/D layout) ----
    const int   n  = n0 + rlo;
    const int   mb = m0 + (lh << 3);
    const float bv = bias[n];
    #pragma unroll
    for (int r = 0; r < 8; ++r)
        hn[(size_t)(mb + r) * RNN_H + n] = f2bf_bits(tanhf(acc0[r] + acc1[r] + bv));
}

// ---------------------------------------------------------------------------
// Final: out[b] = h_T[b,:] . Wf + bf     (tiny: 128x512 dot, h_T in bf16)
// ---------------------------------------------------------------------------
__global__ __launch_bounds__(128) void rnn_final_kernel(
    const unsigned short* __restrict__ hT, const float* __restrict__ Wf,
    const float* __restrict__ bf, float* __restrict__ out)
{
    int b = threadIdx.x;                      // 0..127
    float s = 0.0f;
    #pragma unroll 8
    for (int k = 0; k < RNN_H; ++k)
        s += bf_bits2f(hT[(size_t)b * RNN_H + k]) * Wf[k];
    out[b] = s + bf[0];
}

// ---------------------------------------------------------------------------
// Host launcher. Workspace layout:
//   [0)        Wt  : 512*256 u16   (256 KB)
//   [256KB)    Ut  : 512*512 u16   (512 KB)
//   [768KB)    h0  : 128*512 u16   (128 KB)
//   [896KB)    h1  : 128*512 u16   (128 KB)
//   [1MB)      xbf : 128*512*256 u16 (32 MB)  -- only if ws_size allows
// ---------------------------------------------------------------------------
extern "C" void kernel_launch(void* const* d_in, const int* in_sizes, int n_in,
                              void* d_out, int out_size, void* d_ws, size_t ws_size,
                              hipStream_t stream) {
    (void)in_sizes; (void)n_in; (void)out_size;

    const float* x  = (const float*)d_in[0];   // [128,512,256]
    const float* W  = (const float*)d_in[1];   // [256,512]
    const float* U  = (const float*)d_in[2];   // [512,512]
    const float* b  = (const float*)d_in[3];   // [512]
    const float* Wf = (const float*)d_in[4];   // [512,1]
    const float* bf = (const float*)d_in[5];   // [1]
    float* out = (float*)d_out;                // [128]

    unsigned short* Wt = (unsigned short*)d_ws;
    unsigned short* Ut = Wt + (size_t)RNN_H * RNN_F;
    unsigned short* h0 = Ut + (size_t)RNN_H * RNN_H;
    unsigned short* h1 = h0 + (size_t)RNN_B * RNN_H;
    unsigned short* xb = h1 + (size_t)RNN_B * RNN_H;

    const size_t XN        = (size_t)RNN_B * RNN_T * RNN_F;   // 16M elements
    const size_t base_need = (size_t)((char*)xb - (char*)d_ws);
    const bool   prex      = ws_size >= base_need + XN * sizeof(unsigned short);

    // Prep: transpose+convert weights to bf16, zero h0.
    rnn_prep_kernel<<<(RNN_H * RNN_F + RNN_H * RNN_H + 255) / 256, 256, 0, stream>>>(
        W, U, Wt, Ut, h0);
    if (prex)
        rnn_xcvt_kernel<<<(int)((XN + 255) / 256), 256, 0, stream>>>(x, xb, (int)XN);

    // Sequential scan: 512 stream-ordered step launches, ping-pong h buffers.
    unsigned short* cur = h0;
    unsigned short* nxt = h1;
    for (int t = 0; t < RNN_T; ++t) {
        if (prex)
            rnn_step_kernel<true><<<RNN_H / 16, 256, 0, stream>>>(
                x, xb, Wt, Ut, b, cur, nxt, t);
        else
            rnn_step_kernel<false><<<RNN_H / 16, 256, 0, stream>>>(
                x, xb, Wt, Ut, b, cur, nxt, t);
        unsigned short* tmp = cur; cur = nxt; nxt = tmp;
    }

    // cur now holds h_T (bf16).
    rnn_final_kernel<<<1, 128, 0, stream>>>(cur, Wf, bf, out);
}
